// MixModel_37692632989965
// MI455X (gfx1250) — matrix operand order repaired and checked by reading, not compiled, not run on hardware
//
#include <hip/hip_runtime.h>

typedef __attribute__((ext_vector_type(16))) _Float16 v16h;
typedef __attribute__((ext_vector_type(8)))  _Float16 v8h;
typedef __attribute__((ext_vector_type(8)))  float    v8f;

// f16 weight-fragment workspace layout (units: halves)
#define W1F_OFF    0        // [8 ntile][32 lane][16 e]            = 4096
#define W2AF_OFF   4096     // [4 ntile][4 kk][32 lane][16 e]      = 8192
#define W2BF_OFF   12288    // [5 ntile][4 kk][32 lane][16 e]      = 10240
#define FRAG_TOTAL 22528

#define WAVES 4

// Branch-free tanh: hardware v_tanh_f32 if the toolchain exposes it,
// else exp2/fast-div form (v_exp_f32 + v_rcp_f32, no branches).
__device__ __forceinline__ float tanh_fast(float x) {
#if __has_builtin(__builtin_amdgcn_tanhf)
    return __builtin_amdgcn_tanhf(x);
#else
    // tanh(x) = 1 - 2/(e^{2x}+1);  e^{2x} = 2^{x * 2*log2(e)}
    float e = __builtin_amdgcn_exp2f(x * 2.88539008177792681472f);
    return 1.0f - __fdividef(2.0f, e + 1.0f);
#endif
}

// ---------------------------------------------------------------------------
// One-time weight repack: fp32 row-major -> f16 WMMA B-fragments.
// B 32x16 f16 layout (wave32): col = lane&15, e<8 -> K = (lane>>4)*8 + e; e>=8 -> K = 16 + (lane>>4)*8 + (e-8).
// ---------------------------------------------------------------------------
__global__ void pack_weights(const float* __restrict__ W1,
                             const float* __restrict__ W2a,
                             const float* __restrict__ W2b,
                             _Float16* __restrict__ wf)
{
    int idx = blockIdx.x * blockDim.x + threadIdx.x;
    if (idx >= FRAG_TOTAL) return;
    float val = 0.f;
    if (idx < W2AF_OFF) {                       // W1: 18x128, K padded to 32
        int f = idx;
        int nt   = f >> 9;
        int lane = (f >> 4) & 31;
        int e    = f & 15;
        int col  = nt * 16 + (lane & 15);
        int K    = ((lane >> 4) << 3) + (e & 7) + ((e >> 3) << 4);
        if (K < 18) val = W1[K * 128 + col];
    } else if (idx < W2BF_OFF) {                // W2a: 128x54, N padded to 64
        int f = idx - W2AF_OFF;
        int nt   = f >> 11;
        int kk   = (f >> 9) & 3;
        int lane = (f >> 4) & 31;
        int e    = f & 15;
        int col  = nt * 16 + (lane & 15);
        int K    = kk * 32 + ((lane >> 4) << 3) + (e & 7) + ((e >> 3) << 4);
        if (col < 54) val = W2a[K * 54 + col];
    } else {                                    // W2b: 128x72, N padded to 80
        int f = idx - W2BF_OFF;
        int nt   = f >> 11;
        int kk   = (f >> 9) & 3;
        int lane = (f >> 4) & 31;
        int e    = f & 15;
        int col  = nt * 16 + (lane & 15);
        int K    = kk * 32 + ((lane >> 4) << 3) + (e & 7) + ((e >> 3) << 4);
        if (col < 72) val = W2b[K * 72 + col];
    }
    wf[idx] = (_Float16)val;
}

// ---------------------------------------------------------------------------
// Fused MLP + regression kernel. 4 waves / block, 16 samples / wave.
// ---------------------------------------------------------------------------
__global__ __launch_bounds__(32 * WAVES) void mixmodel_main(
    const float* __restrict__ u,
    const float* __restrict__ reg1, const float* __restrict__ reg2,
    const float* __restrict__ b1,  const float* __restrict__ b2a,
    const float* __restrict__ b2b,
    const _Float16* __restrict__ wf,
    float* __restrict__ out, int Nrows)
{
    __shared__ float    uS [WAVES][16][40];   // u tile, fp32
    __shared__ _Float16 hS [WAVES][2048];     // h, f16, column-major [col*16+row]
    __shared__ float    o0S[WAVES][16][56];   // layer-2a result
    __shared__ float    o1S[WAVES][16][80];   // layer-2b result

    const int wave   = threadIdx.x >> 5;
    const int lane   = threadIdx.x & 31;
    const int m      = lane & 15;             // matrix row / col within tile
    const int halfid = lane >> 4;             // 0 or 1 (lane half)
    const int kb     = halfid * 8;            // A-fragment K base
    const long base  = ((long)blockIdx.x * WAVES + wave) * 16;

    // -------- stage 1: cooperative HBM -> LDS load of 16x36 u tile --------
    #pragma unroll
    for (int i = 0; i < 18; ++i) {
        int f = i * 32 + lane;
        int r = f / 36, c = f - r * 36;
        long row = base + r;
        uS[wave][r][c] = (row < Nrows) ? u[row * 36 + c] : 0.f;
    }
    __syncthreads();

    // -------- stage 2: build A fragment (even cols of u; K=18 padded to 32)
    // A 16x32 f16 layout: row M = lane&15; e<8 -> K = kb+e; e>=8 -> K = 16+kb+(e-8)
    v16h a1;
    #pragma unroll
    for (int e = 0; e < 8; ++e)
        a1[e] = (_Float16)uS[wave][m][2 * (kb + e)];
    #pragma unroll
    for (int e = 8; e < 16; ++e) {
        int k = 16 + kb + (e - 8);
        a1[e] = (k < 18) ? (_Float16)uS[wave][m][2 * k] : (_Float16)0.f;
    }

    // -------- stage 3: layer 1 GEMM (16x18 @ 18x128), bias, tanh ---------
    #pragma unroll
    for (int nt = 0; nt < 8; ++nt) {
        v16h b = *(const v16h*)(wf + W1F_OFF + (nt * 32 + lane) * 16);
        v8f c = {};
        c = __builtin_amdgcn_wmma_f32_16x16x32_f16(false, a1, false, b,
                                                   (short)0, c, false, false);
        float bias = b1[nt * 16 + m];
        v8h hv;
        #pragma unroll
        for (int v = 0; v < 8; ++v) hv[v] = (_Float16)tanh_fast(c[v] + bias);
        // D tile: lane owns col nt*16+m, rows halfid*8..halfid*8+7 -> one b128
        *(v8h*)&hS[wave][(nt * 16 + m) * 16 + halfid * 8] = hv;
    }
    __syncthreads();

    // -------- stage 4: reload h in A-fragment order (4 chunks of K=32) ----
    v16h ha[4];
    #pragma unroll
    for (int kk = 0; kk < 4; ++kk) {
        #pragma unroll
        for (int e = 0; e < 16; ++e) {
            int K = kk * 32 + ((e < 8) ? (kb + e) : (16 + kb + (e - 8)));
            ha[kk][e] = hS[wave][K * 16 + m];
        }
    }

    // -------- stage 5: layer 2a (16x128 @ 128x54) -------------------------
    #pragma unroll
    for (int nt = 0; nt < 4; ++nt) {
        v8f c = {};
        #pragma unroll
        for (int kk = 0; kk < 4; ++kk) {
            v16h b = *(const v16h*)(wf + W2AF_OFF + ((nt * 4 + kk) * 32 + lane) * 16);
            c = __builtin_amdgcn_wmma_f32_16x16x32_f16(false, ha[kk], false, b,
                                                       (short)0, c, false, false);
        }
        int col = nt * 16 + m;
        if (col < 54) {
            float bias = b2a[col];
            #pragma unroll
            for (int v = 0; v < 8; ++v)
                o0S[wave][halfid * 8 + v][col] = c[v] + bias;
        }
    }

    // -------- stage 6: layer 2b (16x128 @ 128x72) -------------------------
    #pragma unroll
    for (int nt = 0; nt < 5; ++nt) {
        v8f c = {};
        #pragma unroll
        for (int kk = 0; kk < 4; ++kk) {
            v16h b = *(const v16h*)(wf + W2BF_OFF + ((nt * 4 + kk) * 32 + lane) * 16);
            c = __builtin_amdgcn_wmma_f32_16x16x32_f16(false, ha[kk], false, b,
                                                       (short)0, c, false, false);
        }
        int col = nt * 16 + m;
        if (col < 72) {
            float bias = b2b[col];
            #pragma unroll
            for (int v = 0; v < 8; ++v)
                o1S[wave][halfid * 8 + v][col] = c[v] + bias;
        }
    }
    __syncthreads();

    // -------- stage 7: per-sample regression + combine + store ------------
    float r1[6], r2[4];
    #pragma unroll
    for (int i = 0; i < 6; ++i) r1[i] = reg1[i];
    #pragma unroll
    for (int i = 0; i < 4; ++i) r2[i] = reg2[i];

    const int  r    = m;            // sample row inside tile
    const long grow = base + r;
    const bool valid = grow < Nrows;

    #pragma unroll
    for (int t = 0; t < 9; ++t) {
        int j  = halfid * 9 + t;    // index pair j -> (even 2j, odd 2j+1)
        int c0 = 2 * j;
        float ue  = uS[wave][r][c0];
        float uo  = uS[wave][r][c0 + 1];
        float um1 = uS[wave][r][(c0 + 35) % 36];  // u[2j-1]
        float um2 = uS[wave][r][(c0 + 34) % 36];  // u[2j-2]
        float up2 = uS[wave][r][(c0 + 2) % 36];   // u[2j+2]
        float up3 = uS[wave][r][(c0 + 3) % 36];   // u[2j+3]

        float re = r1[0] + r1[1]*ue + r1[2]*uo + r1[3]*(um2*um1)
                 + r1[4]*(um1*up2) + r1[5]*(ue*uo);
        float ro = r2[0] + r2[1]*uo + r2[2]*(um1*ue) + r2[3]*(ue*up2);

        float o0a = o0S[wave][r][j*3+0], o0b = o0S[wave][r][j*3+1],
              o0c = o0S[wave][r][j*3+2];
        float o1a = o1S[wave][r][j*4+0], o1b = o1S[wave][r][j*4+1],
              o1c = o1S[wave][r][j*4+2], o1d = o1S[wave][r][j*4+3];

        float oe = o0a + o0b*um1 + o0c*uo + re;
        float oo = o1a + o1b*um1 + o1c*uo + o1d*up3 + ro;

        if (valid) {
            float2 pr = make_float2(oe, oo);
            *(float2*)(out + grow * 36 + c0) = pr;   // 8B coalesced store
        }
    }
}

// ---------------------------------------------------------------------------
extern "C" void kernel_launch(void* const* d_in, const int* in_sizes, int n_in,
                              void* d_out, int out_size, void* d_ws, size_t ws_size,
                              hipStream_t stream)
{
    const float* u    = (const float*)d_in[1];
    const float* reg1 = (const float*)d_in[2];
    const float* reg2 = (const float*)d_in[3];
    const float* W1   = (const float*)d_in[4];
    const float* b1   = (const float*)d_in[5];
    const float* W2a  = (const float*)d_in[6];
    const float* b2a  = (const float*)d_in[7];
    const float* W2b  = (const float*)d_in[8];
    const float* b2b  = (const float*)d_in[9];
    float*    out = (float*)d_out;
    _Float16* wf  = (_Float16*)d_ws;      // 45 KB of f16 fragments
    const int Nrows = in_sizes[1] / 36;

    pack_weights<<<(FRAG_TOTAL + 255) / 256, 256, 0, stream>>>(W1, W2a, W2b, wf);

    int tiles  = (Nrows + 15) / 16;
    int blocks = (tiles + WAVES - 1) / WAVES;
    mixmodel_main<<<blocks, 32 * WAVES, 0, stream>>>(u, reg1, reg2, b1, b2a, b2b,
                                                     wf, out, Nrows);
}
